// PCENLayer_4818953306241
// MI455X (gfx1250) — matrix operand order repaired and checked by reading, not compiled, not run on hardware
//
#include <hip/hip_runtime.h>
#include <hip/hip_bf16.h>

typedef __attribute__((ext_vector_type(16))) _Float16 v16h;
typedef __attribute__((ext_vector_type(8)))  float    v8f;

#define FLOOR_EPS 1e-6f

// Hardware transcendentals (base-2): v_log_f32 / v_exp_f32
#define LOG2F(x) __builtin_amdgcn_logf(x)
#define EXP2F(x) __builtin_amdgcn_exp2f(x)

// Problem shape (fixed by the reference)
constexpr int Bb = 64, Cc = 128, Tt = 8000;
// Time chunking: TC*NCHUNK == T; HALO warmup gives (0.96)^448 ~ 1e-8 error.
constexpr int TC     = 800;   // multiple of 16
constexpr int HALO   = 448;   // multiple of 16
constexpr int NCHUNK = Tt / TC;   // 10
constexpr int NBG    = Bb / 16;   // 4 batch groups of 16

union U16h { v16h v; _Float16 h[16]; };
union U8f  { v8f  v; float    f[8];  };

__device__ __forceinline__ v16h load_bfrag(const float* __restrict__ xrow,
                                           int tt, int hi, _Float16 carry)
{
    // B fragment (32x16 f16): lanes 0-15 carry K-rows 0..15 (x tile),
    // lanes 16-31 carry K-rows 16..31 (row 16 = EMA carry, rest = 0).
    U16h Bf;
    if (!hi) {
        const float4* p = reinterpret_cast<const float4*>(xrow + tt);
        float4 f0 = p[0], f1 = p[1], f2 = p[2], f3 = p[3];
        __builtin_prefetch(xrow + tt + 16, 0, 1);   // global_prefetch_b8
        Bf.h[0]  = (_Float16)f0.x; Bf.h[1]  = (_Float16)f0.y;
        Bf.h[2]  = (_Float16)f0.z; Bf.h[3]  = (_Float16)f0.w;
        Bf.h[4]  = (_Float16)f1.x; Bf.h[5]  = (_Float16)f1.y;
        Bf.h[6]  = (_Float16)f1.z; Bf.h[7]  = (_Float16)f1.w;
        Bf.h[8]  = (_Float16)f2.x; Bf.h[9]  = (_Float16)f2.y;
        Bf.h[10] = (_Float16)f2.z; Bf.h[11] = (_Float16)f2.w;
        Bf.h[12] = (_Float16)f3.x; Bf.h[13] = (_Float16)f3.y;
        Bf.h[14] = (_Float16)f3.z; Bf.h[15] = (_Float16)f3.w;
    } else {
        #pragma unroll
        for (int e = 0; e < 16; ++e) Bf.h[e] = (_Float16)0.f;
        Bf.h[0] = carry;
    }
    return Bf.v;
}

__global__ __launch_bounds__(32)
void pcen_wmma_kernel(const float* __restrict__ x,
                      const float* __restrict__ alpha,
                      const float* __restrict__ delta,
                      const float* __restrict__ root,
                      const float* __restrict__ ema_w,
                      float* __restrict__ out)
{
    const int lane = threadIdx.x;     // wave32
    const int n    = lane & 15;       // batch-in-group / A-row index
    const int hi   = lane >> 4;       // lane half

    int bid = blockIdx.x;
    const int k = bid % NCHUNK; bid /= NCHUNK;   // time chunk
    const int g = bid % NBG;    bid /= NBG;      // batch group
    const int c = bid;                           // channel (uniform -> s_load params)

    // Per-channel parameters (scalar loads; uniform across the wave)
    const float w    = fminf(fmaxf(ema_w[c], 0.f), 1.f);
    const float om   = 1.f - w;
    const float na   = -fminf(alpha[c], 1.f);    // negated exponent: fold the divide
    const float dl   = delta[c];
    const float r    = fmaxf(root[c], 1.f);
    const float oor  = 1.f / r;
    const float dpow = EXP2F(oor * LOG2F(dl));   // delta^(1/root)

    // ---- Build constant A fragment (16x32 f16), row m = output time i ----
    // A[i][j] = w*(1-w)^(i-j) for j<=i<16 ; A[i][16] = (1-w)^(i+1) ; else 0.
    const int m = n;
    U16h A;
    #pragma unroll
    for (int e = 0; e < 16; ++e) {
        const int K = hi * 16 + e;
        float v = 0.f;
        if (K < 16)       { if (K <= m) v = w * __powf(om, (float)(m - K)); }
        else if (K == 16) { v = __powf(om, (float)(m + 1)); }
        A.h[e] = (_Float16)v;
    }

    const size_t row  = (size_t)((g * 16 + n) * Cc + c);
    const float* xrow = x   + row * Tt;
    float*       orow = out + row * Tt;

    const int t0     = k * TC;
    const int tstart = (k == 0) ? 0 : (t0 - HALO);

    // Carry init: s[-1] := x[tstart]  (exact for chunk 0: gives ema[0]=x[0])
    _Float16 carry = (_Float16)xrow[tstart];
    const v8f cz = {};

    // ---- Halo warm-up: EMA only, no stores ----
    for (int tt = tstart; tt < t0; tt += 16) {
        v16h Bv = load_bfrag(xrow, tt, hi, carry);
        U8f D;
        D.v = __builtin_amdgcn_wmma_f32_16x16x32_f16(
                  false, A.v, false, Bv, (short)0, cz, false, false);
        carry = (_Float16)D.f[7];        // s[15][n] lives in hi lanes, VGPR 7
    }

    // ---- Main loop: EMA + elementwise PCEN + store ----
    for (int tt = t0; tt < t0 + TC; tt += 16) {
        v16h Bv = load_bfrag(xrow, tt, hi, carry);
        U8f D;
        D.v = __builtin_amdgcn_wmma_f32_16x16x32_f16(
                  false, A.v, false, Bv, (short)0, cz, false, false);
        carry = (_Float16)D.f[7];

        // D layout: this lane owns batch n, times tt + 8*hi + rr (rr=0..7)
        const float4* q = reinterpret_cast<const float4*>(xrow + tt + 8 * hi);
        float4 xa = q[0], xb = q[1];
        float xv[8] = { xa.x, xa.y, xa.z, xa.w, xb.x, xb.y, xb.z, xb.w };
        float res[8];
        #pragma unroll
        for (int rr = 0; rr < 8; ++rr) {
            const float s    = D.f[rr];
            const float pinv = EXP2F(na * LOG2F(FLOOR_EPS + s));  // (eps+ema)^-a
            const float qv   = __builtin_fmaf(xv[rr], pinv, dl);  // x*pinv + delta
            res[rr] = EXP2F(oor * LOG2F(qv)) - dpow;              // ^(1/r) - d^(1/r)
        }
        float4* o = reinterpret_cast<float4*>(orow + tt + 8 * hi);
        o[0] = make_float4(res[0], res[1], res[2], res[3]);
        o[1] = make_float4(res[4], res[5], res[6], res[7]);
    }
}

extern "C" void kernel_launch(void* const* d_in, const int* in_sizes, int n_in,
                              void* d_out, int out_size, void* d_ws, size_t ws_size,
                              hipStream_t stream) {
    const float* x     = (const float*)d_in[0];
    const float* alpha = (const float*)d_in[1];
    const float* delta = (const float*)d_in[2];
    const float* root  = (const float*)d_in[3];
    const float* ema_w = (const float*)d_in[4];
    float* out = (float*)d_out;

    const int nblocks = Cc * NBG * NCHUNK;   // 128 * 4 * 10 = 5120 waves
    pcen_wmma_kernel<<<nblocks, 32, 0, stream>>>(x, alpha, delta, root, ema_w, out);
}
